// NodeProcessor_1090921693351
// MI455X (gfx1250) — matrix operand order repaired and checked by reading, not compiled, not run on hardware
//
#include <hip/hip_runtime.h>
#include <hip/hip_bf16.h>
#include <cstdint>

typedef __bf16 bf16;
typedef bf16 bf16x16 __attribute__((ext_vector_type(16)));
typedef bf16 bf16x8  __attribute__((ext_vector_type(8)));
typedef float f32x8  __attribute__((ext_vector_type(8)));
typedef int   v4i    __attribute__((ext_vector_type(4)));

#define N_NODES 4608
#define DIM 512
#define NEDGE (N_NODES * 32)          // 147456
#define NETOT (NEDGE + N_NODES)       // 152064 (edges + self loops)
#define NLAYER 6
#define NCEDGE 223488                 // 48 graphs * C(97,2)
#define NPG 96
#define THREADS 256

#define AS1 __attribute__((address_space(1)))
#define AS3 __attribute__((address_space(3)))

#if defined(__has_builtin)
#if __has_builtin(__builtin_amdgcn_global_load_async_to_lds_b128) && \
    __has_builtin(__builtin_amdgcn_s_wait_asynccnt)
#define USE_ASYNC_LDS 1
#endif
#endif
#ifndef USE_ASYNC_LDS
#define USE_ASYNC_LDS 0
#endif

// ---------------- small helpers ----------------
__device__ __forceinline__ unsigned f2ord(float f) {
  unsigned u = __float_as_uint(f);
  return (u & 0x80000000u) ? ~u : (u | 0x80000000u);
}
__device__ __forceinline__ float ord2f(unsigned u) {
  return __uint_as_float((u & 0x80000000u) ? (u ^ 0x80000000u) : ~u);
}
__device__ __forceinline__ float warp_sum(float v) {
#pragma unroll
  for (int o = 16; o > 0; o >>= 1) v += __shfl_xor(v, o, 32);
  return v;
}
__device__ __forceinline__ f32x8 wmma_bf16(bf16x16 a, bf16x16 b, f32x8 c) {
  return __builtin_amdgcn_wmma_f32_16x16x32_bf16(false, a, false, b, (short)0,
                                                 c, false, false);
}

// 16-byte panel fetch into LDS: async DMA if available, else load+ds_store.
__device__ __forceinline__ void panel_fetch16(const bf16* g, bf16* l) {
#if USE_ASYNC_LDS
  __builtin_amdgcn_global_load_async_to_lds_b128(
      (AS1 v4i*)(uintptr_t)g,
      (AS3 v4i*)(uint32_t)(uintptr_t)l, 0, 0);
#else
  *(bf16x8*)l = *(const bf16x8*)g;
#endif
}
__device__ __forceinline__ void panel_wait() {
#if USE_ASYNC_LDS
  __builtin_amdgcn_s_wait_asynccnt(0);
#endif
  __syncthreads();
}

// ---------------- x_proj: h = x @ Wp + bp ----------------
__global__ void k_xproj(const float* __restrict__ x, const float* __restrict__ w,
                        const float* __restrict__ b, float* __restrict__ h,
                        bf16* __restrict__ hbf) {
  int idx = blockIdx.x * THREADS + threadIdx.x;  // n*512 + d
  int n = idx >> 9, d = idx & 511;
  float acc = b[d];
#pragma unroll
  for (int k = 0; k < 16; ++k) acc += x[n * 16 + k] * w[k * 512 + d];
  h[idx] = acc;
  hbf[idx] = (bf16)acc;
}

// ---------------- convert + transpose weight to bf16 col-major ----------------
__global__ void k_wt(const float* __restrict__ w, bf16* __restrict__ wt) {
  int idx = blockIdx.x * THREADS + threadIdx.x;  // k*512 + n
  int k = idx >> 9, n = idx & 511;
  wt[n * 512 + k] = (bf16)w[idx];
}

// ---- WMMA GEMM: block = 256 rows x 64 cols, 8 waves stacked in M.
// ---- Shared 64x32 bf16 weight panel double-buffered in LDS (async staged).
// ---- Each wave: 2 M-subtiles x 4 N-subtiles = 8 WMMA per K-step.
__global__ __launch_bounds__(THREADS)
void k_gemm_bias(const bf16* __restrict__ A, const bf16* __restrict__ Bt,
                 const float* __restrict__ bias, float* __restrict__ C,
                 int M, int Nc, int K) {
  __shared__ bf16 sB[2][64 * 32];  // [buf][panel row][32 K-elems] = 4KB each
  const int tid = threadIdx.x;
  const int wv = tid >> 5, lane = tid & 31;
  const int half = lane >> 4, lrow = lane & 15;
  const int nBlk = Nc >> 6;
  const int bm = blockIdx.x / nBlk;
  const int ncol0 = (blockIdx.x % nBlk) << 6;
  const int row0 = bm * 256 + wv * 32;
  const int prow = tid >> 2, pch = tid & 3;  // panel fetch: row 0..63, 16B chunk 0..3

  f32x8 acc[2][4] = {};
  const bf16* Ar0 = A + (size_t)(row0 + lrow) * K + (half ? 8 : 0);
  const bf16* Ar1 = Ar0 + (size_t)16 * K;
  const bf16* g0 = Bt + (size_t)(ncol0 + prow) * K + pch * 8;
  bf16* lpanel0 = &sB[0][prow * 32 + pch * 8];
  bf16* lpanel1 = &sB[1][prow * 32 + pch * 8];

  panel_fetch16(g0, lpanel0);
  for (int kk = 0; kk < K; kk += 32) {
    int buf = (kk >> 5) & 1;
    panel_wait();
    if (kk + 32 < K) panel_fetch16(g0 + kk + 32, buf ? lpanel0 : lpanel1);
    bf16x16 a0, a1;
    {
      bf16x8 lo = *(const bf16x8*)(Ar0 + kk);
      bf16x8 hi = *(const bf16x8*)(Ar0 + kk + 16);
#pragma unroll
      for (int i = 0; i < 8; ++i) { a0[i] = lo[i]; a0[8 + i] = hi[i]; }
      lo = *(const bf16x8*)(Ar1 + kk);
      hi = *(const bf16x8*)(Ar1 + kk + 16);
#pragma unroll
      for (int i = 0; i < 8; ++i) { a1[i] = lo[i]; a1[8 + i] = hi[i]; }
    }
#pragma unroll
    for (int s = 0; s < 4; ++s) {
      bf16x8 blo = *(const bf16x8*)&sB[buf][(s * 16 + lrow) * 32 + half * 8];
      bf16x8 bhi = *(const bf16x8*)&sB[buf][(s * 16 + lrow) * 32 + 16 + half * 8];
      bf16x16 b = __builtin_shufflevector(blo, bhi, 0, 1, 2, 3, 4, 5, 6, 7, 8, 9, 10, 11, 12, 13, 14, 15);
      acc[0][s] = wmma_bf16(a0, b, acc[0][s]);
      acc[1][s] = wmma_bf16(a1, b, acc[1][s]);
    }
  }
#pragma unroll
  for (int mb = 0; mb < 2; ++mb)
#pragma unroll
    for (int s = 0; s < 4; ++s) {
      int col = ncol0 + s * 16 + lrow;
      float bv = bias ? bias[col] : 0.0f;
#pragma unroll
      for (int r = 0; r < 8; ++r) {
        int row = row0 + mb * 16 + (half ? 8 : 0) + r;
        C[(size_t)row * Nc + col] = acc[mb][s][r] + bv;
      }
    }
}

// ---- CloudGraph GEMM: rows are cloud edges; A = w_e*(h[a]-h[b]) built on the
// ---- fly (amortized over 4 N-subtiles); LDS-staged weight panel; epilogue
// ---- relu(acc+b1) segment-summed into P1 by edge first-index via atomics.
__global__ __launch_bounds__(THREADS)
void k_cloud_gemm(const float* __restrict__ h, const float* __restrict__ wgt,
                  const int* __restrict__ ea, const int* __restrict__ eb,
                  const bf16* __restrict__ Bt, const float* __restrict__ bias,
                  float* __restrict__ P1) {
  __shared__ bf16 sB[2][64 * 32];
  const int tid = threadIdx.x;
  const int wv = tid >> 5, lane = tid & 31;
  const int half = lane >> 4, lrow = lane & 15;
  const int nBlk = DIM >> 6;  // 8
  const int bm = blockIdx.x / nBlk;
  const int ncol0 = (blockIdx.x % nBlk) << 6;
  const int er = bm * 128 + wv * 16 + lrow;
  const int prow = tid >> 2, pch = tid & 3;

  int na = ea[er], nb = eb[er];
  float we = wgt[er];
  const float* ha = h + (size_t)na * DIM + (half ? 8 : 0);
  const float* hb = h + (size_t)nb * DIM + (half ? 8 : 0);
  const bf16* g0 = Bt + (size_t)(ncol0 + prow) * DIM + pch * 8;
  bf16* lpanel0 = &sB[0][prow * 32 + pch * 8];
  bf16* lpanel1 = &sB[1][prow * 32 + pch * 8];

  f32x8 acc[4] = {};
  panel_fetch16(g0, lpanel0);
  for (int kk = 0; kk < DIM; kk += 32) {
    int buf = (kk >> 5) & 1;
    panel_wait();
    if (kk + 32 < DIM) panel_fetch16(g0 + kk + 32, buf ? lpanel0 : lpanel1);
    bf16x16 a;
#pragma unroll
    for (int i = 0; i < 8; ++i) {
      a[i]     = (bf16)(we * (ha[kk + i]      - hb[kk + i]));
      a[8 + i] = (bf16)(we * (ha[kk + 16 + i] - hb[kk + 16 + i]));
    }
#pragma unroll
    for (int s = 0; s < 4; ++s) {
      bf16x8 blo = *(const bf16x8*)&sB[buf][(s * 16 + lrow) * 32 + half * 8];
      bf16x8 bhi = *(const bf16x8*)&sB[buf][(s * 16 + lrow) * 32 + 16 + half * 8];
      bf16x16 b = __builtin_shufflevector(blo, bhi, 0, 1, 2, 3, 4, 5, 6, 7, 8, 9, 10, 11, 12, 13, 14, 15);
      acc[s] = wmma_bf16(a, b, acc[s]);
    }
  }
#pragma unroll
  for (int s = 0; s < 4; ++s) {
    int col = ncol0 + s * 16 + lrow;
    float bv = bias[col];
#pragma unroll
    for (int r = 0; r < 8; ++r) {
      int row = bm * 128 + wv * 16 + (half ? 8 : 0) + r;
      float v = acc[s][r] + bv;
      v = v > 0.0f ? v : 0.0f;
      atomicAdd(&P1[(size_t)ea[row] * DIM + col], v);
    }
  }
}

// ---------------- GAT attention pieces ----------------
__global__ void k_edge_e(const float* __restrict__ xl, const float* __restrict__ xr,
                         const int* __restrict__ ei, const float* __restrict__ att,
                         float* __restrict__ e) {
  int wave = (blockIdx.x * THREADS + threadIdx.x) >> 5;
  int lane = threadIdx.x & 31;
  if (wave >= NETOT) return;
  int src, dst;
  if (wave < NEDGE) { src = ei[wave]; dst = ei[NEDGE + wave]; }
  else              { src = dst = wave - NEDGE; }
  const float* pl = xl + (size_t)src * DIM;
  const float* pr = xr + (size_t)dst * DIM;
  float s = 0.0f;
#pragma unroll
  for (int i = 0; i < 16; ++i) {
    int c = lane * 16 + i;
    float v = pl[c] + pr[c];
    v = v > 0.0f ? v : 0.2f * v;  // leaky_relu(0.2)
    s += v * att[c];
  }
  s = warp_sum(s);
  if (lane == 0) e[wave] = s;
}

__global__ void k_init_seg(unsigned* __restrict__ mord, float* __restrict__ den) {
  int i = blockIdx.x * THREADS + threadIdx.x;
  if (i < N_NODES) { mord[i] = f2ord(-3.0e38f); den[i] = 0.0f; }
}

__global__ void k_segmax(const float* __restrict__ e, const int* __restrict__ ei,
                         unsigned* __restrict__ mord) {
  int t = blockIdx.x * THREADS + threadIdx.x;
  if (t >= NETOT) return;
  int dst = (t < NEDGE) ? ei[NEDGE + t] : (t - NEDGE);
  atomicMax(&mord[dst], f2ord(e[t]));
}

__global__ void k_expsum(float* __restrict__ e, const int* __restrict__ ei,
                         const unsigned* __restrict__ mord, float* __restrict__ den) {
  int t = blockIdx.x * THREADS + threadIdx.x;
  if (t >= NETOT) return;
  int dst = (t < NEDGE) ? ei[NEDGE + t] : (t - NEDGE);
  float ew = __expf(e[t] - ord2f(mord[dst]));
  e[t] = ew;
  atomicAdd(&den[dst], ew);
}

__global__ void k_fill_bias(float* __restrict__ out, const float* __restrict__ bias) {
  int idx = blockIdx.x * THREADS + threadIdx.x;
  out[idx] = bias[idx & 511];
}

__global__ void k_scatter(const float* __restrict__ ew, const float* __restrict__ den,
                          const float* __restrict__ xl, const int* __restrict__ ei,
                          float* __restrict__ out) {
  int wave = (blockIdx.x * THREADS + threadIdx.x) >> 5;
  int lane = threadIdx.x & 31;
  if (wave >= NETOT) return;
  int src, dst;
  if (wave < NEDGE) { src = ei[wave]; dst = ei[NEDGE + wave]; }
  else              { src = dst = wave - NEDGE; }
  float alpha = ew[wave] / den[dst];
  const float* pl = xl + (size_t)src * DIM;
  float* po = out + (size_t)dst * DIM;
#pragma unroll
  for (int i = 0; i < 16; ++i) {
    int c = lane * 16 + i;
    atomicAdd(&po[c], alpha * pl[c]);
  }
}

// ---------------- whole-matrix ("graph") LayerNorm ----------------
__global__ void k_zero(float* __restrict__ p, int n) {
  int i = blockIdx.x * blockDim.x + threadIdx.x;
  if (i < n) p[i] = 0.0f;
}

__global__ void k_stats(const float* __restrict__ x, int n, float* __restrict__ sums) {
  __shared__ float ss[THREADS], sq[THREADS];
  float s = 0.0f, q = 0.0f;
  for (int i = blockIdx.x * THREADS + threadIdx.x; i < n; i += gridDim.x * THREADS) {
    float v = x[i]; s += v; q += v * v;
  }
  ss[threadIdx.x] = s; sq[threadIdx.x] = q;
  __syncthreads();
  for (int o = THREADS / 2; o > 0; o >>= 1) {
    if (threadIdx.x < o) { ss[threadIdx.x] += ss[threadIdx.x + o]; sq[threadIdx.x] += sq[threadIdx.x + o]; }
    __syncthreads();
  }
  if (threadIdx.x == 0) { atomicAdd(&sums[0], ss[0]); atomicAdd(&sums[1], sq[0]); }
}

__global__ void k_graphln(const float* __restrict__ x, const float* __restrict__ sums,
                          const float* __restrict__ g, const float* __restrict__ b,
                          float* __restrict__ y, bf16* __restrict__ ybf) {
  int idx = blockIdx.x * THREADS + threadIdx.x;
  const float cnt = (float)((size_t)N_NODES * DIM);
  float mu = sums[0] / cnt;
  float var = sums[1] / cnt - mu * mu;
  var = var > 0.0f ? var : 0.0f;
  float sd = sqrtf(var);
  int d = idx & 511;
  float v = (x[idx] - mu) / (sd + 1e-5f) * g[d] + b[d];
  v = v > 0.0f ? v : 0.0f;  // relu
  y[idx] = v;
  ybf[idx] = (bf16)v;
}

// ---------------- CloudGraph ----------------
__global__ void k_cloudw(const float* __restrict__ xyz, const int* __restrict__ cea,
                         const int* __restrict__ ceb, float* __restrict__ w) {
  int t = blockIdx.x * THREADS + threadIdx.x;
  if (t >= NCEDGE) return;
  int a3 = cea[t] * 3, b3 = ceb[t] * 3;
  float dx = xyz[a3] - xyz[b3];
  float dy = xyz[a3 + 1] - xyz[b3 + 1];
  float dz = xyz[a3 + 2] - xyz[b3 + 2];
  float sq = dx * dx + dy * dy + dz * dz;
  float dist = sq > 0.0f ? sqrtf(sq) : 0.0f;
  w[t] = __expf(-dist);
}

// p2: per-node sum over its triu neighbors (j >= i within graph), no atomics
__global__ void k_p2(const float* __restrict__ xyz, const float* __restrict__ w3,
                     float* __restrict__ p2) {
  int node = blockIdx.x;                 // 4608 blocks
  int g = node / NPG, li = node % NPG;
  int c0 = threadIdx.x * 2;              // 256 threads * 2 cols
  float acc0 = 0.0f, acc1 = 0.0f;
  float ax = xyz[node * 3], ay = xyz[node * 3 + 1], az = xyz[node * 3 + 2];
  for (int j = li; j < NPG; ++j) {
    int nj = g * NPG + j;
    float dx = ax - xyz[nj * 3];
    float dy = ay - xyz[nj * 3 + 1];
    float dz = az - xyz[nj * 3 + 2];
    float v0 = dx * w3[c0] + dy * w3[512 + c0] + dz * w3[1024 + c0];
    float v1 = dx * w3[c0 + 1] + dy * w3[512 + c0 + 1] + dz * w3[1024 + c0 + 1];
    acc0 += v0 > 0.0f ? v0 : 0.0f;
    acc1 += v1 > 0.0f ? v1 : 0.0f;
  }
  p2[(size_t)node * DIM + c0] = acc0;
  p2[(size_t)node * DIM + c0 + 1] = acc1;
}

// column-wise batchnorm: one block per column
__global__ void k_colbn(const float* __restrict__ x, const float* __restrict__ g,
                        const float* __restrict__ b, float* __restrict__ y) {
  int col = blockIdx.x;  // 512
  __shared__ float ss[THREADS], sq[THREADS];
  float s = 0.0f, q = 0.0f;
  for (int r = threadIdx.x; r < N_NODES; r += THREADS) {
    float v = x[(size_t)r * DIM + col]; s += v; q += v * v;
  }
  ss[threadIdx.x] = s; sq[threadIdx.x] = q;
  __syncthreads();
  for (int o = THREADS / 2; o > 0; o >>= 1) {
    if (threadIdx.x < o) { ss[threadIdx.x] += ss[threadIdx.x + o]; sq[threadIdx.x] += sq[threadIdx.x + o]; }
    __syncthreads();
  }
  __shared__ float smu, sinv;
  if (threadIdx.x == 0) {
    float mu = ss[0] / (float)N_NODES;
    float var = sq[0] / (float)N_NODES - mu * mu;
    var = var > 0.0f ? var : 0.0f;
    smu = mu;
    sinv = 1.0f / sqrtf(var + 1e-5f);
  }
  __syncthreads();
  float mu = smu, inv = sinv, gg = g[col], bb = b[col];
  for (int r = threadIdx.x; r < N_NODES; r += THREADS) {
    size_t i = (size_t)r * DIM + col;
    y[i] = (x[i] - mu) * inv * gg + bb;
  }
}

// row LayerNorm of p1 + fuse: F = 2h + xg + ln1(p1) + bn2, plus bf16 copy
__global__ void k_rowln_fuse(const float* __restrict__ p1, const float* __restrict__ g,
                             const float* __restrict__ b, const float* __restrict__ h,
                             const float* __restrict__ xg, const float* __restrict__ bn2,
                             float* __restrict__ F, bf16* __restrict__ Fbf) {
  int row = blockIdx.x;  // 4608
  __shared__ float ss[THREADS], sq[THREADS];
  int c0 = threadIdx.x * 2;
  size_t base = (size_t)row * DIM;
  float v0 = p1[base + c0], v1 = p1[base + c0 + 1];
  ss[threadIdx.x] = v0 + v1;
  sq[threadIdx.x] = v0 * v0 + v1 * v1;
  __syncthreads();
  for (int o = THREADS / 2; o > 0; o >>= 1) {
    if (threadIdx.x < o) { ss[threadIdx.x] += ss[threadIdx.x + o]; sq[threadIdx.x] += sq[threadIdx.x + o]; }
    __syncthreads();
  }
  float mu = ss[0] / (float)DIM;
  float var = sq[0] / (float)DIM - mu * mu;
  var = var > 0.0f ? var : 0.0f;
  float inv = 1.0f / sqrtf(var + 1e-5f);
#pragma unroll
  for (int t = 0; t < 2; ++t) {
    int c = c0 + t;
    float ln1 = (p1[base + c] - mu) * inv * g[c] + b[c];
    float f = 2.0f * h[base + c] + xg[base + c] + ln1 + bn2[base + c];
    F[base + c] = f;
    Fbf[base + c] = (bf16)f;
  }
}

// ---------------- host orchestration ----------------
static inline size_t alignup(size_t x) { return (x + 255) & ~(size_t)255; }

extern "C" void kernel_launch(void* const* d_in, const int* in_sizes, int n_in,
                              void* d_out, int out_size, void* d_ws, size_t ws_size,
                              hipStream_t stream) {
  const float* x       = (const float*)d_in[0];
  const int*   ei      = (const int*)d_in[1];
  const float* xyz     = (const float*)d_in[3];
  const int*   cei     = (const int*)d_in[4];
  const float* xpw     = (const float*)d_in[5];
  const float* xpb     = (const float*)d_in[6];
  const float* gwl     = (const float*)d_in[7];
  const float* gbl     = (const float*)d_in[8];
  const float* gwr     = (const float*)d_in[9];
  const float* gbr     = (const float*)d_in[10];
  const float* gatt    = (const float*)d_in[11];
  const float* gbias   = (const float*)d_in[12];
  const float* glng    = (const float*)d_in[13];
  const float* glnb    = (const float*)d_in[14];
  const float* cgxyzw  = (const float*)d_in[15];
  const float* cgbng   = (const float*)d_in[16];
  const float* cgbnb   = (const float*)d_in[17];
  const float* cglin1w = (const float*)d_in[18];
  const float* cglin1b = (const float*)d_in[19];
  const float* cglng   = (const float*)d_in[20];
  const float* cglnb   = (const float*)d_in[21];
  const float* linw    = (const float*)d_in[22];
  const float* linb    = (const float*)d_in[23];
  const float* bng     = (const float*)d_in[24];
  const float* bnb     = (const float*)d_in[25];
  float* out = (float*)d_out;

  const size_t ND = (size_t)N_NODES * DIM;
  char* ws = (char*)d_ws;
  size_t off = 0;
  auto carve = [&](size_t bytes) -> void* { void* p = ws + off; off = alignup(off + bytes); return p; };

  float*    h    = (float*)carve(ND * 4);
  bf16*     hbf  = (bf16*)carve(ND * 2);
  float*    bufA = (float*)carve(ND * 4);   // GAT aggregation output / xg
  float*    bufB = (float*)carve(ND * 4);   // normalized f32
  bf16*     xgbf = (bf16*)carve(ND * 2);
  float*    xl   = (float*)carve(ND * 4);
  float*    xr   = (float*)carve(ND * 4);
  float*    ebuf = (float*)carve((size_t)NETOT * 4);
  float*    den  = (float*)carve((size_t)N_NODES * 4);
  unsigned* mord = (unsigned*)carve((size_t)N_NODES * 4);
  bf16*     wtl  = (bf16*)carve((size_t)512 * 512 * 2);
  bf16*     wtr  = (bf16*)carve((size_t)512 * 512 * 2);
  float*    wcl  = (float*)carve((size_t)NCEDGE * 4);
  float*    p1   = (float*)carve(ND * 4);
  float*    p2   = (float*)carve(ND * 4);
  float*    F    = (float*)carve(ND * 4);
  bf16*     Fbf  = (bf16*)carve(ND * 2);
  float*    pre  = (float*)carve(ND * 4);
  float*    sums = (float*)carve(8 * 4);

  const int  gND       = (int)(ND / THREADS);                        // 9216
  const int  gemmBlks  = (N_NODES / 256) * (DIM / 64);               // 144
  const int  edgeWBlks = (NETOT * 32 + THREADS - 1) / THREADS;       // 19008
  const int  edgeTBlks = (NETOT + THREADS - 1) / THREADS;            // 594
  const int  cgemmBlks = (NCEDGE / 128) * (DIM / 64);                // 13968
  const int* cea = cei;
  const int* ceb = cei + NCEDGE;

  // h = x @ x_proj_w + b (also bf16 copy)
  k_xproj<<<gND, THREADS, 0, stream>>>(x, xpw, xpb, h, hbf);

  // ---- GATv2 stack ----
  const bf16* cur = hbf;
  for (int l = 0; l < NLAYER; ++l) {
    k_wt<<<1024, THREADS, 0, stream>>>(gwl + (size_t)l * 512 * 512, wtl);
    k_wt<<<1024, THREADS, 0, stream>>>(gwr + (size_t)l * 512 * 512, wtr);
    k_gemm_bias<<<gemmBlks, THREADS, 0, stream>>>(cur, wtl, gbl + l * 512, xl, N_NODES, DIM, DIM);
    k_gemm_bias<<<gemmBlks, THREADS, 0, stream>>>(cur, wtr, gbr + l * 512, xr, N_NODES, DIM, DIM);
    k_init_seg<<<(N_NODES + THREADS - 1) / THREADS, THREADS, 0, stream>>>(mord, den);
    k_edge_e<<<edgeWBlks, THREADS, 0, stream>>>(xl, xr, ei, gatt + l * 512, ebuf);
    k_segmax<<<edgeTBlks, THREADS, 0, stream>>>(ebuf, ei, mord);
    k_expsum<<<edgeTBlks, THREADS, 0, stream>>>(ebuf, ei, mord, den);
    k_fill_bias<<<gND, THREADS, 0, stream>>>(bufA, gbias + l * 512);
    k_scatter<<<edgeWBlks, THREADS, 0, stream>>>(ebuf, den, xl, ei, bufA);
    if (l < NLAYER - 1) {
      k_zero<<<1, 32, 0, stream>>>(sums, 2);
      k_stats<<<1024, THREADS, 0, stream>>>(bufA, (int)ND, sums);
      k_graphln<<<gND, THREADS, 0, stream>>>(bufA, sums, glng + l * 512, glnb + l * 512, bufB, xgbf);
      cur = xgbf;
    }
  }
  // final GAT output (xg) lives in bufA

  // ---- CloudGraph ----
  k_wt<<<1024, THREADS, 0, stream>>>(cglin1w, wtl);
  k_cloudw<<<(NCEDGE + THREADS - 1) / THREADS, THREADS, 0, stream>>>(xyz, cea, ceb, wcl);
  k_zero<<<gND, THREADS, 0, stream>>>(p1, (int)ND);
  k_cloud_gemm<<<cgemmBlks, THREADS, 0, stream>>>(h, wcl, cea, ceb, wtl, cglin1b, p1);
  k_p2<<<N_NODES, THREADS, 0, stream>>>(xyz, cgxyzw, p2);
  k_colbn<<<DIM, THREADS, 0, stream>>>(p2, cgbng, cgbnb, p2);  // in-place bn2
  k_rowln_fuse<<<N_NODES, THREADS, 0, stream>>>(p1, cglng, cglnb, h, bufA, p2, F, Fbf);

  // ---- final GEMM + batchnorm ----
  k_wt<<<1024, THREADS, 0, stream>>>(linw, wtr);
  k_gemm_bias<<<gemmBlks, THREADS, 0, stream>>>(Fbf, wtr, linb, pre, N_NODES, DIM, DIM);
  k_colbn<<<DIM, THREADS, 0, stream>>>(pre, bng, bnb, out);
}
